// Encoder_27290222198965
// MI455X (gfx1250) — matrix operand order repaired and checked by reading, not compiled, not run on hardware
//
#include <hip/hip_runtime.h>
#include <hip/hip_bf16.h>
#include <math.h>

// Problem: N=8, L=2048, H=1024 -> M = N*L = 16384 rows, all GEMMs M x H x H.
#define NB 8
#define LL 2048
#define HH 1024
#define MM (NB * LL)

typedef _Float16 h8   __attribute__((ext_vector_type(8)));
typedef _Float16 v16h __attribute__((ext_vector_type(16)));
typedef _Float16 h4   __attribute__((ext_vector_type(4)));
typedef float    v8f  __attribute__((ext_vector_type(8)));

// ---------- WMMA fragment loaders (v_wmma_f32_16x16x32_f16 layouts) ----------
// A matrix 16x32 f16, contiguous k order used here (not the ISA table order):
//   lanes 0-15 : row M=lane,      K = k0+0..15 contiguous (VGPR0-7)
//   lanes 16-31: row M=lane-16,   K = k0+16..31 contiguous
__device__ __forceinline__ v16h loadA(const _Float16* __restrict__ rowp, int k0, int hi8) {
    const h8* p0 = (const h8*)(rowp + k0 + 2 * hi8);
    const h8* p1 = (const h8*)(rowp + k0 + 2 * hi8 + 8);
    h8 a = *p0;
    h8 b = *p1;
    return __builtin_shufflevector(a, b, 0,1,2,3,4,5,6,7,8,9,10,11,12,13,14,15);
}
// B matrix 32x16 f16, same contiguous k order as loadA (not the ISA table order):
//   lanes 0-15 : col N=lane,      K = k0+0..15 contiguous (VGPR0-7)
//   lanes 16-31: col N=lane-16,   K = k0+16..31 contiguous
// Since B = W^T, column n of B is row n of W (contiguous along K).
__device__ __forceinline__ v16h loadB(const _Float16* __restrict__ rowp, int k0, int hi16) {
    const h8* p0 = (const h8*)(rowp + k0 + hi16);
    h8 a = p0[0];
    h8 b = p0[1];
    return __builtin_shufflevector(a, b, 0,1,2,3,4,5,6,7,8,9,10,11,12,13,14,15);
}

__device__ __forceinline__ v8f wmma_f16(v16h a, v16h b, v8f c) {
    return __builtin_amdgcn_wmma_f32_16x16x32_f16(false, a, false, b, (short)0, c, false, false);
}

// ---------- cast f32 -> f16, 4 elements per thread ----------
__global__ void cast_f32_to_f16_kernel(const float* __restrict__ in,
                                       _Float16* __restrict__ out, int count4) {
    int i = blockIdx.x * blockDim.x + threadIdx.x;
    if (i < count4) {
        float4 v = ((const float4*)in)[i];
        h4 o;
        o[0] = (_Float16)v.x; o[1] = (_Float16)v.y;
        o[2] = (_Float16)v.z; o[3] = (_Float16)v.w;
        ((h4*)out)[i] = o;
    }
}

// ---------- fused q.k rowwise-dot: d[m] = sum_h (xWq^T+bq)*(xWk^T+bk) ----------
// grid.x = MM/16 blocks, 256 threads (8 waves). Wave w owns col tiles w*8..w*8+7.
// k-loop outermost; 16 persistent accumulators (128 VGPRs). B fragments are
// manually double-buffered inside the (fully unrolled) tile loop so tile i+1's
// global_load_b128s are in flight while tile i's WMMAs execute.
__global__ void __launch_bounds__(256, 2)
qkdot_kernel(const _Float16* __restrict__ xh,
             const _Float16* __restrict__ wqh,
             const _Float16* __restrict__ wkh,
             const float* __restrict__ bq,
             const float* __restrict__ bk,
             float* __restrict__ dout) {
    const int lane = threadIdx.x & 31;
    const int wv   = threadIdx.x >> 5;
    const int l16  = lane & 15;
    const int hi8  = (lane >= 16) ? 8 : 0;
    const int hi16 = (lane >= 16) ? 16 : 0;
    const int m0   = blockIdx.x * 16;

    const _Float16* arow   = xh + (size_t)(m0 + l16) * HH;
    const _Float16* bqbase = wqh + (size_t)(wv * 128 + l16) * HH; // wv*8 tiles * 16 rows
    const _Float16* bkbase = wkh + (size_t)(wv * 128 + l16) * HH;

    v8f accq[8], acck[8];
#pragma unroll
    for (int i = 0; i < 8; ++i) { v8f z = {}; accq[i] = z; acck[i] = z; }

#pragma unroll 1
    for (int k0 = 0; k0 < HH; k0 += 32) {
        v16h a    = loadA(arow, k0, hi8);
        v16h bq_c = loadB(bqbase, k0, hi16);
        v16h bk_c = loadB(bkbase, k0, hi16);
#pragma unroll
        for (int i = 0; i < 8; ++i) {
            v16h bq_n, bk_n;
            if (i < 7) {
                bq_n = loadB(bqbase + (size_t)(i + 1) * 16 * HH, k0, hi16);
                bk_n = loadB(bkbase + (size_t)(i + 1) * 16 * HH, k0, hi16);
            }
            accq[i] = wmma_f16(a, bq_c, accq[i]);
            acck[i] = wmma_f16(a, bk_c, acck[i]);
            if (i < 7) { bq_c = bq_n; bk_c = bk_n; }
        }
    }

    float s[8];
#pragma unroll
    for (int r = 0; r < 8; ++r) s[r] = 0.f;
#pragma unroll
    for (int i = 0; i < 8; ++i) {
        const int n0 = (wv * 8 + i) * 16;
        const float bqv = bq[n0 + l16];
        const float bkv = bk[n0 + l16];
#pragma unroll
        for (int r = 0; r < 8; ++r)
            s[r] += (accq[i][r] + bqv) * (acck[i][r] + bkv);
    }

    // reduce across the 16 lanes sharing each row (masks < 16 stay in half)
#pragma unroll
    for (int r = 0; r < 8; ++r) {
        s[r] += __shfl_xor(s[r], 1, 32);
        s[r] += __shfl_xor(s[r], 2, 32);
        s[r] += __shfl_xor(s[r], 4, 32);
        s[r] += __shfl_xor(s[r], 8, 32);
    }

    __shared__ float lds[8][16];
    if (lane == 0) {
#pragma unroll
        for (int r = 0; r < 8; ++r) lds[wv][r] = s[r];
    }
    if (lane == 16) {
#pragma unroll
        for (int r = 0; r < 8; ++r) lds[wv][8 + r] = s[r];
    }
    __syncthreads();
    if (threadIdx.x < 16) {
        float acc = 0.f;
#pragma unroll
        for (int w = 0; w < 8; ++w) acc += lds[w][threadIdx.x];  // fixed order
        dout[m0 + threadIdx.x] = acc;
    }
}

// ---------- per-batch S[n] and colsum[n,l] (deterministic tree reduce) ----------
__global__ void colsum_kernel(const float* __restrict__ d,
                              float* __restrict__ colsum) {
    const int n   = blockIdx.x;
    const int tid = threadIdx.x;
    __shared__ float red[256];

    float local = 0.f;
#pragma unroll
    for (int j = 0; j < LL / 256; ++j) {
        float dd = d[n * LL + tid + j * 256];
        float rt;
        if (dd > 0.f) { float em = expf(-dd); rt = em / (1.f + 2047.f * em); }
        else          { float ep = expf(dd);  rt = 1.f / (ep + 2047.f); }
        local += rt;
    }
    red[tid] = local;
    __syncthreads();
    for (int s2 = 128; s2 > 0; s2 >>= 1) {
        if (tid < s2) red[tid] += red[tid + s2];
        __syncthreads();
    }
    const float S = red[0];
#pragma unroll
    for (int j = 0; j < LL / 256; ++j) {
        int l = tid + j * 256;
        float dd = d[n * LL + l];
        float part;
        if (dd > 0.f) { float em = expf(-dd); part = (1.f - em) / (1.f + 2047.f * em); }
        else          { float ep = expf(dd);  part = (ep - 1.f) / (ep + 2047.f); }
        colsum[n * LL + l] = part + S;
    }
}

// ---------- v GEMM fused with bias + colsum scaling -> f16 contexts ----------
// 2 row-tiles x 4 col-tiles per wave: 8 wmma per 12 b128 loads.
// jobs = (MM/32)*(HH/64) = 8192, 8 waves/block -> 1024 blocks.
__global__ void __launch_bounds__(256, 2)
vgemm_scale_kernel(const _Float16* __restrict__ xh,
                   const _Float16* __restrict__ wvh,
                   const float* __restrict__ bv,
                   const float* __restrict__ colsum,
                   _Float16* __restrict__ ctxh) {
    const int lane = threadIdx.x & 31;
    const int wv   = threadIdx.x >> 5;
    const int l16  = lane & 15;
    const int hi8  = (lane >= 16) ? 8 : 0;
    const int hi16 = (lane >= 16) ? 16 : 0;

    const int job = blockIdx.x * 8 + wv;
    const int mt  = job >> 4;       // 0..511 (32-row tiles)
    const int ng  = job & 15;       // 0..15  (64-col groups)
    const int m0  = mt * 32;
    const int n0  = ng * 64;

    const _Float16* arow0 = xh + (size_t)(m0 + l16) * HH;
    const _Float16* arow1 = arow0 + (size_t)16 * HH;
    const _Float16* brow  = wvh + (size_t)(n0 + l16) * HH;

    v8f acc0[4], acc1[4];
#pragma unroll
    for (int j = 0; j < 4; ++j) { v8f z = {}; acc0[j] = z; acc1[j] = z; }

    v16h a0 = loadA(arow0, 0, hi8);
    v16h a1 = loadA(arow1, 0, hi8);
    v16h b0 = loadB(brow + (size_t)0 * 16 * HH, 0, hi16);
    v16h b1 = loadB(brow + (size_t)1 * 16 * HH, 0, hi16);
    v16h b2 = loadB(brow + (size_t)2 * 16 * HH, 0, hi16);
    v16h b3 = loadB(brow + (size_t)3 * 16 * HH, 0, hi16);

#pragma unroll 1
    for (int k0 = 0; k0 < HH; k0 += 32) {
        const int ks = (k0 + 32 < HH) ? (k0 + 32) : 0;
        v16h a0n = loadA(arow0, ks, hi8);
        v16h a1n = loadA(arow1, ks, hi8);
        v16h b0n = loadB(brow + (size_t)0 * 16 * HH, ks, hi16);
        v16h b1n = loadB(brow + (size_t)1 * 16 * HH, ks, hi16);
        v16h b2n = loadB(brow + (size_t)2 * 16 * HH, ks, hi16);
        v16h b3n = loadB(brow + (size_t)3 * 16 * HH, ks, hi16);

        acc0[0] = wmma_f16(a0, b0, acc0[0]);
        acc1[0] = wmma_f16(a1, b0, acc1[0]);
        acc0[1] = wmma_f16(a0, b1, acc0[1]);
        acc1[1] = wmma_f16(a1, b1, acc1[1]);
        acc0[2] = wmma_f16(a0, b2, acc0[2]);
        acc1[2] = wmma_f16(a1, b2, acc1[2]);
        acc0[3] = wmma_f16(a0, b3, acc0[3]);
        acc1[3] = wmma_f16(a1, b3, acc1[3]);

        a0 = a0n; a1 = a1n; b0 = b0n; b1 = b1n; b2 = b2n; b3 = b3n;
    }

    float cs0[8], cs1[8];
#pragma unroll
    for (int r = 0; r < 8; ++r) {
        cs0[r] = colsum[m0 + r + hi8];
        cs1[r] = colsum[m0 + 16 + r + hi8];
    }

#pragma unroll
    for (int j = 0; j < 4; ++j) {
        const int col = n0 + j * 16 + l16;
        const float bvv = bv[col];
#pragma unroll
        for (int r = 0; r < 8; ++r) {
            const int row0 = m0 + r + hi8;
            const int row1 = row0 + 16;
            ctxh[(size_t)row0 * HH + col] = (_Float16)((acc0[j][r] + bvv) * cs0[r]);
            ctxh[(size_t)row1 * HH + col] = (_Float16)((acc1[j][r] + bvv) * cs1[r]);
        }
    }
}

// ---------- out GEMM: d_out = ctx Wo^T + bo (f32 out) ----------
__global__ void __launch_bounds__(256, 2)
outgemm_kernel(const _Float16* __restrict__ ctxh,
               const _Float16* __restrict__ woh,
               const float* __restrict__ bo,
               float* __restrict__ out) {
    const int lane = threadIdx.x & 31;
    const int wv   = threadIdx.x >> 5;
    const int l16  = lane & 15;
    const int hi8  = (lane >= 16) ? 8 : 0;
    const int hi16 = (lane >= 16) ? 16 : 0;

    const int job = blockIdx.x * 8 + wv;
    const int mt  = job >> 4;
    const int ng  = job & 15;
    const int m0  = mt * 32;
    const int n0  = ng * 64;

    const _Float16* arow0 = ctxh + (size_t)(m0 + l16) * HH;
    const _Float16* arow1 = arow0 + (size_t)16 * HH;
    const _Float16* brow  = woh + (size_t)(n0 + l16) * HH;

    v8f acc0[4], acc1[4];
#pragma unroll
    for (int j = 0; j < 4; ++j) { v8f z = {}; acc0[j] = z; acc1[j] = z; }

    v16h a0 = loadA(arow0, 0, hi8);
    v16h a1 = loadA(arow1, 0, hi8);
    v16h b0 = loadB(brow + (size_t)0 * 16 * HH, 0, hi16);
    v16h b1 = loadB(brow + (size_t)1 * 16 * HH, 0, hi16);
    v16h b2 = loadB(brow + (size_t)2 * 16 * HH, 0, hi16);
    v16h b3 = loadB(brow + (size_t)3 * 16 * HH, 0, hi16);

#pragma unroll 1
    for (int k0 = 0; k0 < HH; k0 += 32) {
        const int ks = (k0 + 32 < HH) ? (k0 + 32) : 0;
        v16h a0n = loadA(arow0, ks, hi8);
        v16h a1n = loadA(arow1, ks, hi8);
        v16h b0n = loadB(brow + (size_t)0 * 16 * HH, ks, hi16);
        v16h b1n = loadB(brow + (size_t)1 * 16 * HH, ks, hi16);
        v16h b2n = loadB(brow + (size_t)2 * 16 * HH, ks, hi16);
        v16h b3n = loadB(brow + (size_t)3 * 16 * HH, ks, hi16);

        acc0[0] = wmma_f16(a0, b0, acc0[0]);
        acc1[0] = wmma_f16(a1, b0, acc1[0]);
        acc0[1] = wmma_f16(a0, b1, acc0[1]);
        acc1[1] = wmma_f16(a1, b1, acc1[1]);
        acc0[2] = wmma_f16(a0, b2, acc0[2]);
        acc1[2] = wmma_f16(a1, b2, acc1[2]);
        acc0[3] = wmma_f16(a0, b3, acc0[3]);
        acc1[3] = wmma_f16(a1, b3, acc1[3]);

        a0 = a0n; a1 = a1n; b0 = b0n; b1 = b1n; b2 = b2n; b3 = b3n;
    }

#pragma unroll
    for (int j = 0; j < 4; ++j) {
        const int col = n0 + j * 16 + l16;
        const float bov = bo[col];
#pragma unroll
        for (int r = 0; r < 8; ++r) {
            const int row0 = m0 + r + hi8;
            const int row1 = row0 + 16;
            out[(size_t)row0 * HH + col] = acc0[j][r] + bov;
            out[(size_t)row1 * HH + col] = acc1[j][r] + bov;
        }
    }
}

extern "C" void kernel_launch(void* const* d_in, const int* in_sizes, int n_in,
                              void* d_out, int out_size, void* d_ws, size_t ws_size,
                              hipStream_t stream) {
    (void)in_sizes; (void)n_in; (void)out_size; (void)ws_size;
    const float* x  = (const float*)d_in[0];
    const float* Wq = (const float*)d_in[1];
    const float* bq = (const float*)d_in[2];
    const float* Wk = (const float*)d_in[3];
    const float* bk = (const float*)d_in[4];
    const float* Wv = (const float*)d_in[5];
    const float* bv = (const float*)d_in[6];
    const float* Wo = (const float*)d_in[7];
    const float* bo = (const float*)d_in[8];
    float* out = (float*)d_out;

    // workspace layout (~76 MB)
    char* ws = (char*)d_ws;
    size_t off = 0;
    auto alloc = [&](size_t bytes) -> void* {
        off = (off + 255) & ~(size_t)255;
        void* p = ws + off;
        off += bytes;
        return p;
    };
    _Float16* xh   = (_Float16*)alloc((size_t)MM * HH * 2);
    _Float16* wqh  = (_Float16*)alloc((size_t)HH * HH * 2);
    _Float16* wkh  = (_Float16*)alloc((size_t)HH * HH * 2);
    _Float16* wvh  = (_Float16*)alloc((size_t)HH * HH * 2);
    _Float16* woh  = (_Float16*)alloc((size_t)HH * HH * 2);
    _Float16* ctxh = (_Float16*)alloc((size_t)MM * HH * 2);
    float*    dbuf = (float*)alloc((size_t)MM * 4);
    float*    csum = (float*)alloc((size_t)MM * 4);

    // casts
    {
        int c4 = MM * HH / 4;
        cast_f32_to_f16_kernel<<<(c4 + 255) / 256, 256, 0, stream>>>(x, xh, c4);
        int w4 = HH * HH / 4;
        cast_f32_to_f16_kernel<<<(w4 + 255) / 256, 256, 0, stream>>>(Wq, wqh, w4);
        cast_f32_to_f16_kernel<<<(w4 + 255) / 256, 256, 0, stream>>>(Wk, wkh, w4);
        cast_f32_to_f16_kernel<<<(w4 + 255) / 256, 256, 0, stream>>>(Wv, wvh, w4);
        cast_f32_to_f16_kernel<<<(w4 + 255) / 256, 256, 0, stream>>>(Wo, woh, w4);
    }

    // fused q.k rowwise dot -> d
    qkdot_kernel<<<MM / 16, 256, 0, stream>>>(xh, wqh, wkh, bq, bk, dbuf);

    // per-batch softmax column sums
    colsum_kernel<<<NB, 256, 0, stream>>>(dbuf, csum);

    // v GEMM + bias + scale -> ctx (f16)
    vgemm_scale_kernel<<<(MM / 32) * (HH / 64) / 8, 256, 0, stream>>>(xh, wvh, bv, csum, ctxh);

    // out GEMM + bias -> f32 output
    outgemm_kernel<<<(MM / 32) * (HH / 64) / 8, 256, 0, stream>>>(ctxh, woh, bo, out);
}
